// MeshGraphNet_9655086482216
// MI455X (gfx1250) — matrix operand order repaired and checked by reading, not compiled, not run on hardware
//
#include <hip/hip_runtime.h>
#include <hip/hip_bf16.h>

typedef __attribute__((ext_vector_type(16))) _Float16 v16h;
typedef __attribute__((ext_vector_type(8)))  float    v8f;

constexpr int HDIM = 128;

// ---------------------------------------------------------------------------
// WMMA fragment helpers (layouts per CDNA5 ISA §7.12.2, wave32)
// ---------------------------------------------------------------------------

// A fragment: 16x32 f16 tile starting at column `kbase` of a row-major LDS
// tile with row stride `stride` (in elements).
// lane<16: M=lane, K = kbase+{0..7} then kbase+{16..23}
// lane>=16: M=lane-16, K = kbase+{8..15} then kbase+{24..31}
__device__ __forceinline__ v16h load_a_frag(const _Float16* lds, int lane,
                                            int kbase, int stride) {
  const int m  = lane & 15;
  const int hi = (lane >> 4) & 1;
  const _Float16* row = lds + m * stride + kbase;
  const int k0 = hi ? 8 : 0;
  const int k1 = hi ? 24 : 16;
  v16h a;
#pragma unroll
  for (int e = 0; e < 8; ++e) a[e] = row[k0 + e];
#pragma unroll
  for (int e = 0; e < 8; ++e) a[8 + e] = row[k1 + e];
  return a;
}

// B fragment: weights pre-swizzled so each lane's 16 f16 values are
// contiguous: packed[((kt*8 + ct)*32 + lane)*16 + e]
__device__ __forceinline__ v16h load_b_frag(const _Float16* packW, int kt,
                                            int ct, int lane) {
  return *(const v16h*)(packW + ((((kt << 3) + ct) << 5) + lane) * 16);
}

// ---------------------------------------------------------------------------
// Fused MLP tile over RT*16 rows: out = LN(relu(A@W1+b1)@W2 + b2) stats.
// A staged in ldsA (RT*16 x K1PAD f16). Pre-LN results left in ldsOut
// (RT*16 x 128 f32); per-row mean in lnS[0..RT*16), rsqrt(var+eps) in
// lnS[RT*16..2*RT*16). Block = 256 threads = 8 waves; wave w owns output
// columns [16w,16w+16) and iterates RT row-subtiles, reusing each B fragment
// RT times (halves L2 weight traffic, doubles wmma density).
// ---------------------------------------------------------------------------
template <int K1PAD, int RT>
__device__ __forceinline__ void mlp_tile(const _Float16* ldsA, _Float16* ldsH,
                                         float* ldsOut, float* lnS,
                                         const _Float16* pW1, const float* b1,
                                         const _Float16* pW2, const float* b2) {
  const int tid  = threadIdx.x;
  const int lane = tid & 31;
  const int wave = tid >> 5;                 // column tile 0..7
  const int col  = (wave << 4) + (lane & 15);
  const int hi   = (lane >> 4) & 1;

  // GEMM1: [RT*16 x K1PAD] @ [K1PAD x 128]
  v8f acc[RT] = {};
  for (int kt = 0; kt < K1PAD / 32; ++kt) {
    v16h b = load_b_frag(pW1, kt, wave, lane);
#pragma unroll
    for (int rt = 0; rt < RT; ++rt) {
      v16h a = load_a_frag(ldsA + rt * 16 * K1PAD, lane, kt * 32, K1PAD);
      acc[rt] = __builtin_amdgcn_wmma_f32_16x16x32_f16(
          false, a, false, b, (short)0, acc[rt], false, false);
    }
  }
  const float bias1 = b1[col];
#pragma unroll
  for (int rt = 0; rt < RT; ++rt)
#pragma unroll
    for (int r = 0; r < 8; ++r) {
      float v = acc[rt][r] + bias1;
      v = v > 0.f ? v : 0.f;                              // relu
      ldsH[(rt * 16 + r + hi * 8) * HDIM + col] = (_Float16)v;
    }
  __syncthreads();

  // GEMM2: [RT*16 x 128] @ [128 x 128]
  v8f acc2[RT] = {};
#pragma unroll
  for (int kt = 0; kt < 4; ++kt) {
    v16h b = load_b_frag(pW2, kt, wave, lane);
#pragma unroll
    for (int rt = 0; rt < RT; ++rt) {
      v16h a = load_a_frag(ldsH + rt * 16 * HDIM, lane, kt * 32, HDIM);
      acc2[rt] = __builtin_amdgcn_wmma_f32_16x16x32_f16(
          false, a, false, b, (short)0, acc2[rt], false, false);
    }
  }
  const float bias2 = b2[col];
#pragma unroll
  for (int rt = 0; rt < RT; ++rt)
#pragma unroll
    for (int r = 0; r < 8; ++r)
      ldsOut[(rt * 16 + r + hi * 8) * HDIM + col] = acc2[rt][r] + bias2;
  __syncthreads();

  // Per-row LayerNorm statistics
  if (tid < 16 * RT) {
    const float* row = ldsOut + tid * HDIM;
    float m = 0.f;
    for (int c = 0; c < HDIM; ++c) m += row[c];
    m *= (1.f / HDIM);
    float v = 0.f;
    for (int c = 0; c < HDIM; ++c) { float d = row[c] - m; v += d * d; }
    v *= (1.f / HDIM);
    lnS[tid] = m;
    lnS[16 * RT + tid] = rsqrtf(v + 1e-5f);
  }
  __syncthreads();
}

// ---------------------------------------------------------------------------
// Weight repack: row-major f32 [Ksrc x 128] -> B-fragment-ordered f16,
// K zero-padded to Kpad (multiple of 32).
// ---------------------------------------------------------------------------
__global__ void repack_kernel(const float* __restrict__ W,
                              _Float16* __restrict__ dst, int Ksrc, int total) {
  int idx = blockIdx.x * 256 + threadIdx.x;
  if (idx >= total) return;
  int e    = idx & 15;
  int lane = (idx >> 4) & 31;
  int ct   = (idx >> 9) & 7;
  int kt   = idx >> 12;
  int col  = (ct << 4) + (lane & 15);
  int k    = kt * 32 + ((e >> 3) << 4) + ((lane >> 4) << 3) + (e & 7);
  dst[idx] = (k < Ksrc) ? (_Float16)W[k * HDIM + col] : (_Float16)0.f;
}

// ---------------------------------------------------------------------------
// Encoders: out = LN(relu(in@W1+b1)@W2+b2)*gamma+beta   (KIN padded to 32)
// 32 rows per block.
// ---------------------------------------------------------------------------
template <int KIN>
__global__ void encode_kernel(const float* __restrict__ in,
                              const _Float16* pW1, const float* b1,
                              const _Float16* pW2, const float* b2,
                              const float* gamma, const float* beta,
                              float* __restrict__ out) {
  __shared__ _Float16 sA[32 * 32];
  __shared__ _Float16 sH[32 * HDIM];
  __shared__ float    sO[32 * HDIM];
  __shared__ float    sLN[64];
  const int tid  = threadIdx.x;
  const int row0 = blockIdx.x * 32;
  for (int idx = tid; idx < 32 * 32; idx += 256) {
    int r = idx >> 5, k = idx & 31;
    float v = (k < KIN) ? in[(row0 + r) * KIN + k] : 0.f;
    sA[idx] = (_Float16)v;
  }
  __syncthreads();
  mlp_tile<32, 2>(sA, sH, sO, sLN, pW1, b1, pW2, b2);
  for (int idx = tid; idx < 32 * HDIM; idx += 256) {
    int r = idx >> 7, c = idx & 127;
    out[(row0 + r) * HDIM + c] =
        (sO[idx] - sLN[r]) * sLN[32 + r] * gamma[c] + beta[c];
  }
}

// ---------------------------------------------------------------------------
// Edge update: x_e += LN-MLP([xv[s], xv[r], x_e, g]); scatter into aggr.
// 32 edges per block.
// ---------------------------------------------------------------------------
__global__ void edge_layer_kernel(const float* __restrict__ xv,
                                  float* __restrict__ xe,
                                  const float* __restrict__ g,
                                  const int* __restrict__ senders,
                                  const int* __restrict__ receivers,
                                  const _Float16* pW1, const float* b1,
                                  const _Float16* pW2, const float* b2,
                                  const float* gamma, const float* beta,
                                  float* __restrict__ aggr) {
  __shared__ _Float16 sA[32 * 512];
  __shared__ _Float16 sH[32 * HDIM];
  __shared__ float    sO[32 * HDIM];
  __shared__ float    sLN[64];
  __shared__ int      sSnd[32], sRcv[32];
  const int tid  = threadIdx.x;
  const int row0 = blockIdx.x * 32;
  if (tid < 32) {
    sSnd[tid] = senders[row0 + tid];
    sRcv[tid] = receivers[row0 + tid];
  }
  __syncthreads();
  for (int idx = tid; idx < 32 * 512; idx += 256) {
    int r = idx >> 9, k = idx & 511;
    float v;
    if (k < 128)      v = xv[sSnd[r] * HDIM + k];
    else if (k < 256) v = xv[sRcv[r] * HDIM + (k - 128)];
    else if (k < 384) v = xe[(row0 + r) * HDIM + (k - 256)];
    else              v = g[k - 384];
    sA[idx] = (_Float16)v;
  }
  __syncthreads();
  mlp_tile<512, 2>(sA, sH, sO, sLN, pW1, b1, pW2, b2);
  for (int idx = tid; idx < 32 * HDIM; idx += 256) {
    int r = idx >> 7, c = idx & 127;
    float v = (sO[idx] - sLN[r]) * sLN[32 + r] * gamma[c] + beta[c];
    float nv = xe[(row0 + r) * HDIM + c] + v;          // residual
    xe[(row0 + r) * HDIM + c] = nv;
    atomicAdd(&aggr[sRcv[r] * HDIM + c], nv);          // segment_sum
  }
}

// ---------------------------------------------------------------------------
// Node update: x_v += LN-MLP([x_v, aggr, g])   32 nodes per block.
// ---------------------------------------------------------------------------
__global__ void node_layer_kernel(float* __restrict__ xv,
                                  const float* __restrict__ aggr,
                                  const float* __restrict__ g,
                                  const _Float16* pW1, const float* b1,
                                  const _Float16* pW2, const float* b2,
                                  const float* gamma, const float* beta) {
  __shared__ _Float16 sA[32 * 384];
  __shared__ _Float16 sH[32 * HDIM];
  __shared__ float    sO[32 * HDIM];
  __shared__ float    sLN[64];
  const int tid  = threadIdx.x;
  const int row0 = blockIdx.x * 32;
  for (int idx = tid; idx < 32 * 384; idx += 256) {
    int r = idx / 384, k = idx % 384;
    float v;
    if (k < 128)      v = xv[(row0 + r) * HDIM + k];
    else if (k < 256) v = aggr[(row0 + r) * HDIM + (k - 128)];
    else              v = g[k - 256];
    sA[idx] = (_Float16)v;
  }
  __syncthreads();
  mlp_tile<384, 2>(sA, sH, sO, sLN, pW1, b1, pW2, b2);
  for (int idx = tid; idx < 32 * HDIM; idx += 256) {
    int r = idx >> 7, c = idx & 127;
    float v = (sO[idx] - sLN[r]) * sLN[32 + r] * gamma[c] + beta[c];
    xv[(row0 + r) * HDIM + c] += v;                    // residual
  }
}

// ---------------------------------------------------------------------------
// Column sums (for global feature): out[c] += sum_r mat[r][c]
// ---------------------------------------------------------------------------
__global__ void colsum_kernel(const float* __restrict__ mat, int rows,
                              float* __restrict__ out) {
  const int c  = threadIdx.x;        // 128 threads
  const int r0 = blockIdx.x * 512;
  const int r1 = min(r0 + 512, rows);
  float s = 0.f;
  for (int r = r0; r < r1; ++r) s += mat[r * HDIM + c];
  atomicAdd(&out[c], s);
}

// ---------------------------------------------------------------------------
// Global feature update: g += LN-MLP([sum_v, sum_e, g])  (tiny, scalar fp32)
// ---------------------------------------------------------------------------
__global__ void global_update_kernel(const float* __restrict__ sums,
                                     float* __restrict__ g,
                                     const float* __restrict__ W1,
                                     const float* __restrict__ b1,
                                     const float* __restrict__ W2,
                                     const float* __restrict__ b2,
                                     const float* __restrict__ gamma,
                                     const float* __restrict__ beta) {
  __shared__ float gin[384];
  __shared__ float h[128];
  __shared__ float o[128];
  __shared__ float st[2];
  const int t = threadIdx.x;         // 128 threads
  gin[t]       = sums[t];            // sum over x_v
  gin[128 + t] = sums[128 + t];      // sum over x_e
  gin[256 + t] = g[t];
  __syncthreads();
  float a = b1[t];
  for (int k = 0; k < 384; ++k) a += gin[k] * W1[k * HDIM + t];
  h[t] = a > 0.f ? a : 0.f;
  __syncthreads();
  float b = b2[t];
  for (int k = 0; k < 128; ++k) b += h[k] * W2[k * HDIM + t];
  o[t] = b;
  __syncthreads();
  if (t == 0) {
    float m = 0.f;
    for (int c = 0; c < 128; ++c) m += o[c];
    m *= (1.f / 128.f);
    float v = 0.f;
    for (int c = 0; c < 128; ++c) { float d = o[c] - m; v += d * d; }
    st[0] = m;
    st[1] = rsqrtf(v * (1.f / 128.f) + 1e-5f);
  }
  __syncthreads();
  g[t] += (o[t] - st[0]) * st[1] * gamma[t] + beta[t];
}

// ---------------------------------------------------------------------------
// Decoder: per-node conv1(k=15,s=4,oc=8)+relu -> conv2(k=10,s=1,oc=1)
// One wave per node (8 nodes / 256-thread block).
// ---------------------------------------------------------------------------
__global__ void decoder_kernel(const float* __restrict__ xv,
                               const float* __restrict__ c1w,
                               const float* __restrict__ c1b,
                               const float* __restrict__ c2w,
                               const float* __restrict__ c2b,
                               float* __restrict__ out) {
  __shared__ float sx[8][HDIM];
  __shared__ float sh[8][8 * 29];
  const int tid  = threadIdx.x;
  const int w    = tid >> 5;
  const int lane = tid & 31;
  const int node = blockIdx.x * 8 + w;
#pragma unroll
  for (int i = 0; i < 4; ++i)
    sx[w][lane * 4 + i] = xv[node * HDIM + lane * 4 + i];
  __syncthreads();
  for (int idx = lane; idx < 8 * 29; idx += 32) {
    int oc = idx / 29, t = idx % 29;
    float a = c1b[oc];
#pragma unroll
    for (int j = 0; j < 15; ++j) a += sx[w][4 * t + j] * c1w[oc * 15 + j];
    sh[w][oc * 29 + t] = a > 0.f ? a : 0.f;
  }
  __syncthreads();
  if (lane < 20) {
    float a = c2b[0];
#pragma unroll
    for (int oc = 0; oc < 8; ++oc)
#pragma unroll
      for (int j = 0; j < 10; ++j)
        a += sh[w][oc * 29 + lane + j] * c2w[oc * 10 + j];
    out[node * 20 + lane] = a;
  }
}

// ---------------------------------------------------------------------------
// Host orchestration
// ---------------------------------------------------------------------------
extern "C" void kernel_launch(void* const* d_in, const int* in_sizes, int n_in,
                              void* d_out, int out_size, void* d_ws,
                              size_t ws_size, hipStream_t stream) {
  const float* x         = (const float*)d_in[0];
  const float* edge_attr = (const float*)d_in[1];
  const int*   eidx      = (const int*)d_in[2];
  const float* Wn1  = (const float*)d_in[3];
  const float* bn1  = (const float*)d_in[4];
  const float* Wn2  = (const float*)d_in[5];
  const float* bn2  = (const float*)d_in[6];
  const float* gnln = (const float*)d_in[7];
  const float* bnln = (const float*)d_in[8];
  const float* We1  = (const float*)d_in[9];
  const float* be1  = (const float*)d_in[10];
  const float* We2  = (const float*)d_in[11];
  const float* be2  = (const float*)d_in[12];
  const float* geln = (const float*)d_in[13];
  const float* beln = (const float*)d_in[14];
  const float* lEW1 = (const float*)d_in[15];
  const float* lEb1 = (const float*)d_in[16];
  const float* lEW2 = (const float*)d_in[17];
  const float* lEb2 = (const float*)d_in[18];
  const float* lEg  = (const float*)d_in[19];
  const float* lEbt = (const float*)d_in[20];
  const float* lVW1 = (const float*)d_in[21];
  const float* lVb1 = (const float*)d_in[22];
  const float* lVW2 = (const float*)d_in[23];
  const float* lVb2 = (const float*)d_in[24];
  const float* lVg  = (const float*)d_in[25];
  const float* lVbt = (const float*)d_in[26];
  const float* lGW1 = (const float*)d_in[27];
  const float* lGb1 = (const float*)d_in[28];
  const float* lGW2 = (const float*)d_in[29];
  const float* lGb2 = (const float*)d_in[30];
  const float* lGg  = (const float*)d_in[31];
  const float* lGbt = (const float*)d_in[32];
  const float* c1w  = (const float*)d_in[33];
  const float* c1b  = (const float*)d_in[34];
  const float* c2w  = (const float*)d_in[35];
  const float* c2b  = (const float*)d_in[36];

  const int N = in_sizes[0] / 12;    // 20000
  const int E = in_sizes[1] / 4;     // 160000
  const int L = 4;
  const int* senders   = eidx;
  const int* receivers = eidx + E;

  // Workspace carve-out
  char*  base = (char*)d_ws;
  size_t off  = 0;
  auto take = [&](size_t bytes) -> char* {
    off = (off + 255) & ~(size_t)255;
    char* p = base + off;
    off += bytes;
    return p;
  };
  float* xv   = (float*)take((size_t)N * HDIM * 4);
  float* xe   = (float*)take((size_t)E * HDIM * 4);
  float* aggr = (float*)take((size_t)N * HDIM * 4);
  float* gbuf = (float*)take(HDIM * 4);
  float* sums = (float*)take(2 * HDIM * 4);
  _Float16* pWn1 = (_Float16*)take(32 * 128 * 2);
  _Float16* pWn2 = (_Float16*)take(128 * 128 * 2);
  _Float16* pWe1 = (_Float16*)take(32 * 128 * 2);
  _Float16* pWe2 = (_Float16*)take(128 * 128 * 2);
  _Float16 *pEW1[4], *pEW2[4], *pVW1[4], *pVW2[4];
  for (int l = 0; l < L; ++l) {
    pEW1[l] = (_Float16*)take(512 * 128 * 2);
    pEW2[l] = (_Float16*)take(128 * 128 * 2);
    pVW1[l] = (_Float16*)take(384 * 128 * 2);
    pVW2[l] = (_Float16*)take(128 * 128 * 2);
  }

  auto repack = [&](const float* W, _Float16* dst, int Ksrc, int Kpad) {
    int total = Kpad * 128;
    repack_kernel<<<(total + 255) / 256, 256, 0, stream>>>(W, dst, Ksrc, total);
  };
  repack(Wn1, pWn1, 12, 32);
  repack(Wn2, pWn2, 128, 128);
  repack(We1, pWe1, 4, 32);
  repack(We2, pWe2, 128, 128);
  for (int l = 0; l < L; ++l) {
    repack(lEW1 + (size_t)l * 512 * 128, pEW1[l], 512, 512);
    repack(lEW2 + (size_t)l * 128 * 128, pEW2[l], 128, 128);
    repack(lVW1 + (size_t)l * 384 * 128, pVW1[l], 384, 384);
    repack(lVW2 + (size_t)l * 128 * 128, pVW2[l], 128, 128);
  }

  hipMemsetAsync(gbuf, 0, HDIM * sizeof(float), stream);

  // Encoders (32 rows per block)
  encode_kernel<12><<<N / 32, 256, 0, stream>>>(x, pWn1, bn1, pWn2, bn2, gnln,
                                                bnln, xv);
  encode_kernel<4><<<E / 32, 256, 0, stream>>>(edge_attr, pWe1, be1, pWe2, be2,
                                               geln, beln, xe);

  // Message-passing layers
  for (int l = 0; l < L; ++l) {
    hipMemsetAsync(aggr, 0, (size_t)N * HDIM * sizeof(float), stream);
    edge_layer_kernel<<<E / 32, 256, 0, stream>>>(
        xv, xe, gbuf, senders, receivers, pEW1[l], lEb1 + l * HDIM, pEW2[l],
        lEb2 + l * HDIM, lEg + l * HDIM, lEbt + l * HDIM, aggr);
    node_layer_kernel<<<N / 32, 256, 0, stream>>>(
        xv, aggr, gbuf, pVW1[l], lVb1 + l * HDIM, pVW2[l], lVb2 + l * HDIM,
        lVg + l * HDIM, lVbt + l * HDIM);
    hipMemsetAsync(sums, 0, 2 * HDIM * sizeof(float), stream);
    colsum_kernel<<<(N + 511) / 512, 128, 0, stream>>>(xv, N, sums);
    colsum_kernel<<<(E + 511) / 512, 128, 0, stream>>>(xe, E, sums + HDIM);
    global_update_kernel<<<1, 128, 0, stream>>>(
        sums, gbuf, lGW1 + (size_t)l * 384 * 128, lGb1 + l * HDIM,
        lGW2 + (size_t)l * 128 * 128, lGb2 + l * HDIM, lGg + l * HDIM,
        lGbt + l * HDIM);
  }

  // Decoder
  decoder_kernel<<<N / 8, 256, 0, stream>>>(xv, c1w, c1b, c2w, c2b,
                                            (float*)d_out);
}
